// EfficientAttention_1692217114782
// MI455X (gfx1250) — matrix operand order repaired and checked by reading, compiled, not run on hardware
//
#include <hip/hip_runtime.h>
#include <hip/hip_bf16.h>
#include <stdint.h>

// ---------------------------------------------------------------------------
// MI455X (gfx1250) multi-head attention forward.
//   DIM=1024, HEADS=16, HEAD_DIM=64, B=2, S=2048.
// Compute-bound (~60 GFLOP vs ~3us of HBM traffic) -> all matmuls run through
// v_wmma_f32_16x16x32_f16 with f32 accumulation.
// Attention K/V tiles are staged into LDS by the Tensor Data Mover
// (tensor_load_to_lds, double buffered, TENSORcnt-synchronized); LDS tiles are
// padded via the D# pad feature so every fragment gather is a 16B-aligned
// ds_load_b128.
// ---------------------------------------------------------------------------

typedef __attribute__((ext_vector_type(16))) _Float16 v16h;
typedef __attribute__((ext_vector_type(8)))  float    v8f;
typedef __attribute__((ext_vector_type(4)))  unsigned int v4u;
typedef __attribute__((ext_vector_type(8)))  int      v8i;
typedef __attribute__((ext_vector_type(4)))  int      v4i;

union F16Frag {
    v16h     v;
    uint32_t u[8];
    _Float16 h[16];
};

union H2Pack {
    uint32_t u;
    _Float16 h[2];
};

__device__ __forceinline__ v8f wmma_f16(v16h a, v16h b, v8f c) {
    // D = A(16x32 f16) x B(32x16 f16) + C(16x16 f32)
    return __builtin_amdgcn_wmma_f32_16x16x32_f16(
        /*neg_a=*/false, a, /*neg_b=*/false, b,
        /*c_mod=*/(short)0, c, /*reuse_a=*/false, /*reuse_b=*/false);
}

#define DIM      1024
#define HEADS    16
#define HDIM     64
#define SEQ      2048
#define BATCH    2
#define MTOT     (BATCH * SEQ)   // 4096

#if __has_builtin(__builtin_amdgcn_tensor_load_to_lds) && \
    __has_builtin(__builtin_amdgcn_s_wait_tensorcnt)
#define ATTN_USE_TDM 1
#else
#define ATTN_USE_TDM 0
#endif

#if ATTN_USE_TDM
// ---------------------------------------------------------------------------
// Issue one TDM 2D tile load (f16 elements) global -> LDS.
//   tile0/tile1  : tile dims (dim0 = contiguous elements)
//   ten0/ten1    : tensor dims (for OOB clamping)
//   stride0      : tensor dim0 stride in elements
//   padi/pada    : D# pad_interval / pad_amount codes (LDS row padding)
// D# layout per CDNA5 ISA ch.8 (group0: 128b, group1: 256b; groups 2/3 unused
// for 2D tiles). This toolchain exposes the 6-arg builtin:
//   (uint32x4 g0, int32x8 g1, int32x4 g2, int32x4 g3, int32x8 g4, i32 cpol)
// ---------------------------------------------------------------------------
__device__ __forceinline__ void tdm_load_2d_f16(
    const void* gptr, uint32_t lds_off,
    uint32_t tile0, uint32_t tile1,
    uint32_t ten0, uint32_t ten1,
    uint64_t stride0, uint32_t padi, uint32_t pada)
{
    uint64_t ga = (uint64_t)gptr;
    v4u g0;
    g0[0] = 1u;                                            // count=1, user D#
    g0[1] = lds_off;                                       // lds_addr (bytes)
    g0[2] = (uint32_t)ga;                                  // global_addr[31:0]
    g0[3] = (uint32_t)((ga >> 32) & 0x01FFFFFFu) | (2u << 30); // addr[56:32], type=2

    v8i g1;
    g1[0] = (int)((1u << 16)            // data_size = 1 -> 2 bytes
                | (1u << 20)            // pad_enable
                | (padi << 22)          // pad_interval code
                | (pada << 25));        // pad_amount code
    g1[1] = (int)((ten0 & 0xFFFFu) << 16);                 // tensor_dim0[15:0]
    g1[2] = (int)((ten0 >> 16) | ((ten1 & 0xFFFFu) << 16));// dim0[31:16], dim1[15:0]
    g1[3] = (int)((ten1 >> 16) | (tile0 << 16));           // dim1[31:16], tile_dim0
    g1[4] = (int)(tile1 & 0xFFFFu);                        // tile_dim1 (tile_dim2=0)
    g1[5] = (int)(uint32_t)stride0;                        // dim0_stride[31:0]
    g1[6] = (int)(uint32_t)((stride0 >> 32) & 0xFFFFu);    // dim0_stride[47:32]
    g1[7] = 0;

    v4i g2 = {0, 0, 0, 0};
    v4i g3 = {0, 0, 0, 0};
    v8i g4 = {0, 0, 0, 0, 0, 0, 0, 0};
    __builtin_amdgcn_tensor_load_to_lds(g0, g1, g2, g3, g4, 0);
}
#endif

// ---------------------------------------------------------------------------
// Kernel 1: QKV projection.  out = (x @ W + b) [* 0.125 for Q]
// grid = (MTOT/64, DIM/128, 3), block = 256 (8 waves).
// Wave tile 32x32, workgroup tile 64(M) x 128(N), K-step 32.
// fp32 -> f16 conversion while staging into LDS; all LDS staging stores are
// packed dwords, all fragment reads are 16B-aligned ds_load_b128.
// Q/K written f16 [B,H,S,dh]; V written f16 [B,H,dh,S] (transposed for TDM).
// ---------------------------------------------------------------------------
__global__ __launch_bounds__(256) void qkv_proj_kernel(
    const float* __restrict__ x,
    const float* __restrict__ Wq, const float* __restrict__ bq,
    const float* __restrict__ Wk, const float* __restrict__ bk,
    const float* __restrict__ Wv, const float* __restrict__ bv,
    _Float16* __restrict__ Qo, _Float16* __restrict__ Ko, _Float16* __restrict__ Vo)
{
    __shared__ _Float16 sA [64][40];   // 64 M x 32 K   (pad 40 -> 80B rows)
    __shared__ _Float16 sBt[128][40];  // 128 N x 32 K  transposed (80B rows)

    const int t    = threadIdx.x;
    const int lane = t & 31;
    const int w    = t >> 5;
    const int wm   = w & 1;
    const int wn   = w >> 1;
    const int hi16 = lane >> 4;
    const int ln   = lane & 15;

    const float* W; const float* bias; _Float16* O; float scale;
    if (blockIdx.z == 0)      { W = Wq; bias = bq; O = Qo; scale = 0.125f; } // dh^-0.5 folded in
    else if (blockIdx.z == 1) { W = Wk; bias = bk; O = Ko; scale = 1.0f; }
    else                      { W = Wv; bias = bv; O = Vo; scale = 1.0f; }
    const bool transposeOut = (blockIdx.z == 2);

    const int m0blk = blockIdx.x * 64;
    const int n0blk = blockIdx.y * 128;

    v8f acc[2][2] = {};

    for (int kt = 0; kt < DIM; kt += 32) {
        // stage A tile 64x32: float2 load, packed-dword LDS store
        #pragma unroll
        for (int i = 0; i < 4; ++i) {
            int e = i * 256 + t;             // dword slot 0..1023
            int r = e >> 4, c2 = e & 15;
            float2 xv = *(const float2*)&x[(size_t)(m0blk + r) * DIM + kt + c2 * 2];
            H2Pack p; p.h[0] = (_Float16)xv.x; p.h[1] = (_Float16)xv.y;
            *(uint32_t*)&sA[r][c2 * 2] = p.u;
        }
        // stage W tile transposed: pack rows k,k+1 of column n -> one dword
        #pragma unroll
        for (int i = 0; i < 8; ++i) {
            int e  = i * 256 + t;            // dword slot 0..2047
            int k2 = e >> 7, c = e & 127;    // k-pair, col n (coalesced in n)
            float w0 = W[(size_t)(kt + 2 * k2)     * DIM + n0blk + c];
            float w1 = W[(size_t)(kt + 2 * k2 + 1) * DIM + n0blk + c];
            H2Pack p; p.h[0] = (_Float16)w0; p.h[1] = (_Float16)w1;
            *(uint32_t*)&sBt[c][k2 * 2] = p.u;
        }
        __syncthreads();

        F16Frag a[2], b[2];
        #pragma unroll
        for (int f = 0; f < 2; ++f) {
            int row = wm * 32 + f * 16 + ln;
            #pragma unroll
            for (int v = 0; v < 8; ++v) {
                int k0 = ((v < 4) ? 0 : 16) + hi16 * 8 + (v & 3) * 2;
                a[f].u[v] = *(const uint32_t*)&sA[row][k0];
            }
            int col = wn * 32 + f * 16 + ln;
            #pragma unroll
            for (int v = 0; v < 8; ++v) {
                int k = ((v < 4) ? 0 : 16) + hi16 * 8 + (v & 3) * 2;   // table order -> 2x b128
                b[f].u[v] = *(const uint32_t*)&sBt[col][k];
            }
        }
        #pragma unroll
        for (int i = 0; i < 2; ++i)
            #pragma unroll
            for (int j = 0; j < 2; ++j)
                acc[i][j] = wmma_f16(a[i].v, b[j].v, acc[i][j]);
        __syncthreads();
    }

    // epilogue: +bias, (Q: *0.125), f16 scatter
    #pragma unroll
    for (int i = 0; i < 2; ++i) {
        #pragma unroll
        for (int j = 0; j < 2; ++j) {
            int n  = n0blk + wn * 32 + j * 16 + ln;
            float bn = bias[n];
            int hcol = n >> 6, d = n & 63;
            #pragma unroll
            for (int v = 0; v < 8; ++v) {
                int m = m0blk + wm * 32 + i * 16 + hi16 * 8 + v;
                int bb = m >> 11, s = m & (SEQ - 1);
                float val = (acc[i][j][v] + bn) * scale;
                size_t idx = transposeOut
                    ? (((size_t)bb * HEADS + hcol) * HDIM + d) * SEQ + s    // V^T: [B,H,dh,S]
                    : (((size_t)bb * HEADS + hcol) * SEQ + s) * HDIM + d;   // Q,K: [B,H,S,dh]
                O[idx] = (_Float16)val;
            }
        }
    }
}

// ---------------------------------------------------------------------------
// Kernel 2: flash attention.  grid = (SEQ/128, B*H), block = 256 (8 waves).
// Each wave owns 16 query rows; keys processed 32 at a time.
// K [S,dh] and V^T [dh,S] tiles staged by TDM (double buffered) with LDS row
// padding chosen so fragment gathers are 16B-aligned ds_load_b128.
// ---------------------------------------------------------------------------
__global__ __launch_bounds__(256) void attn_kernel(
    const _Float16* __restrict__ Q,
    const _Float16* __restrict__ Km,
    const _Float16* __restrict__ Vt,
    float* __restrict__ out)
{
    __shared__ _Float16 sK [2][32][72];   // 32 keys x 64 dh, 144B rows (TDM pad)
    __shared__ _Float16 sVt[2][64][40];   // 64 dh x 32 keys,  80B rows (TDM pad)
    __shared__ _Float16 sP [8][16][40];   // per-wave P relayout (16x32, 80B rows)

    const int t    = threadIdx.x;
    const int lane = t & 31;
    const int w    = t >> 5;
    const int hi16 = lane >> 4;
    const int ln   = lane & 15;

    const int bh = blockIdx.y;
    const int b  = bh >> 4, h = bh & 15;
    const int q0 = blockIdx.x * 128 + w * 16;

    const _Float16* Qp  = Q  + (size_t)bh * SEQ * HDIM;
    const _Float16* Kp  = Km + (size_t)bh * SEQ * HDIM;
    const _Float16* Vtp = Vt + (size_t)bh * HDIM * SEQ;

    // Q fragments for K-dim 0..31 and 32..63 (loaded once, already *0.125)
    F16Frag aq[2];
    {
        int row = q0 + ln;
        #pragma unroll
        for (int f = 0; f < 2; ++f)
            #pragma unroll
            for (int v = 0; v < 8; ++v) {
                int k0 = f * 32 + ((v < 4) ? 0 : 16) + hi16 * 8 + (v & 3) * 2;
                aq[f].u[v] = *(const uint32_t*)(Qp + (size_t)row * HDIM + k0);
            }
    }

    float rm[8], rl[8];
    #pragma unroll
    for (int v = 0; v < 8; ++v) { rm[v] = -__builtin_inff(); rl[v] = 0.0f; }
    v8f o[4] = {};

#if ATTN_USE_TDM
    // Prologue: wave 0 launches the DMA for tile 0 into buffer 0.
    if (w == 0) {
        // K tile: 32 rows x 64 elems; pad 128B rows by 16B -> 144B (72 f16)
        tdm_load_2d_f16(Kp, (uint32_t)(uintptr_t)&sK[0][0][0],
                        /*tile*/ 64, 32, /*tensor*/ 64, SEQ,
                        /*stride0*/ 64, /*padi(32DW)*/ 4, /*pada(4DW)*/ 3);
        // V^T tile: 64 rows x 32 elems; pad 64B rows by 16B -> 80B (40 f16)
        tdm_load_2d_f16(Vtp, (uint32_t)(uintptr_t)&sVt[0][0][0],
                        /*tile*/ 32, 64, /*tensor*/ SEQ, HDIM,
                        /*stride0*/ SEQ, /*padi(16DW)*/ 3, /*pada(4DW)*/ 3);
    }
#endif

    for (int jt = 0; jt < SEQ; jt += 32) {
        const int buf = (jt >> 5) & 1;

#if ATTN_USE_TDM
        if (w == 0) __builtin_amdgcn_s_wait_tensorcnt(0);  // in-flight tile done
        __syncthreads();                                   // publish to all waves
        if (w == 0 && (jt + 32) < SEQ) {                   // prefetch next tile
            int nb = buf ^ 1, nj = jt + 32;
            tdm_load_2d_f16(Kp + (size_t)nj * HDIM, (uint32_t)(uintptr_t)&sK[nb][0][0],
                            64, 32, 64, SEQ, 64, 4, 3);
            tdm_load_2d_f16(Vtp + nj, (uint32_t)(uintptr_t)&sVt[nb][0][0],
                            32, 64, SEQ, HDIM, SEQ, 3, 3);
        }
#else
        // Fallback: cooperative vector-load staging (u32 = 2 f16)
        #pragma unroll
        for (int i = 0; i < 4; ++i) {
            int e = i * 256 + t;
            { int r = e >> 5, c2 = e & 31;
              *(uint32_t*)&sK[buf][r][c2 * 2] =
                  *(const uint32_t*)(Kp + (size_t)(jt + r) * HDIM + c2 * 2); }
            { int r = e >> 4, c2 = e & 15;
              *(uint32_t*)&sVt[buf][r][c2 * 2] =
                  *(const uint32_t*)(Vtp + (size_t)r * SEQ + jt + c2 * 2); }
        }
        __syncthreads();
#endif

        // scores: S = Q(16x64) @ K_tile^T -> two 16x16 f32 fragments
        v8f sfr[2];
        #pragma unroll
        for (int g = 0; g < 2; ++g) {
            v8f s = {};
            int key = g * 16 + ln;
            #pragma unroll
            for (int kk = 0; kk < 2; ++kk) {
                F16Frag bk;
                #pragma unroll
                for (int v = 0; v < 8; ++v) {
                    int k = kk * 32 + ((v < 4) ? 0 : 16) + hi16 * 8 + (v & 3) * 2;   // table order -> 2x b128
                    bk.u[v] = *(const uint32_t*)&sK[buf][key][k];
                }
                s = wmma_f16(aq[kk].v, bk.v, s);
            }
            sfr[g] = s;
        }

        // ---- online softmax (row reductions across the 16-lane group) ----
        float mx[8];
        #pragma unroll
        for (int v = 0; v < 8; ++v) mx[v] = fmaxf(sfr[0][v], sfr[1][v]);
        #pragma unroll
        for (int mask = 1; mask <= 8; mask <<= 1)
            #pragma unroll
            for (int v = 0; v < 8; ++v)
                mx[v] = fmaxf(mx[v], __shfl_xor(mx[v], mask, 32));

        float tsum[8];
        #pragma unroll
        for (int v = 0; v < 8; ++v) {
            float mnew  = fmaxf(rm[v], mx[v]);
            float alpha = __expf(rm[v] - mnew);
            float p0 = __expf(sfr[0][v] - mnew);
            float p1 = __expf(sfr[1][v] - mnew);
            sfr[0][v] = p0; sfr[1][v] = p1;
            rm[v] = mnew;
            rl[v] *= alpha;
            tsum[v] = p0 + p1;
            #pragma unroll
            for (int f = 0; f < 4; ++f) o[f][v] *= alpha;
        }
        #pragma unroll
        for (int mask = 1; mask <= 8; mask <<= 1)
            #pragma unroll
            for (int v = 0; v < 8; ++v)
                tsum[v] += __shfl_xor(tsum[v], mask, 32);
        #pragma unroll
        for (int v = 0; v < 8; ++v) rl[v] += tsum[v];

        // relayout P (C-format f32 -> A-format f16) through per-wave LDS
        #pragma unroll
        for (int g = 0; g < 2; ++g)
            #pragma unroll
            for (int v = 0; v < 8; ++v)
                sP[w][hi16 * 8 + v][g * 16 + ln] = (_Float16)sfr[g][v];
        asm volatile("s_wait_dscnt 0" ::: "memory");   // wave-local LDS RAW fence

        F16Frag ap;
        #pragma unroll
        for (int v = 0; v < 8; ++v) {
            int k0 = ((v < 4) ? 0 : 16) + hi16 * 8 + (v & 3) * 2;
            ap.u[v] = *(const uint32_t*)&sP[w][ln][k0];
        }

        // O += P(16x32) @ V_tile(32x64): B fragments straight from V^T rows
        #pragma unroll
        for (int f = 0; f < 4; ++f) {
            F16Frag bv;
            int col = f * 16 + ln;                         // output dh column
            #pragma unroll
            for (int v = 0; v < 8; ++v) {
                int k = ((v < 4) ? 0 : 16) + hi16 * 8 + (v & 3) * 2;   // table order -> 2x b128
                bv.u[v] = *(const uint32_t*)&sVt[buf][col][k];
            }
            o[f] = wmma_f16(ap.v, bv.v, o[f]);
        }

#if !ATTN_USE_TDM
        __syncthreads();   // protect single-buffer staging
#endif
    }

    // epilogue: one reciprocal per row, store f32 [B, S, H*dh]
    float inv[8];
    #pragma unroll
    for (int v = 0; v < 8; ++v) inv[v] = 1.0f / rl[v];
    #pragma unroll
    for (int f = 0; f < 4; ++f) {
        int n = f * 16 + ln;
        #pragma unroll
        for (int v = 0; v < 8; ++v) {
            int s = q0 + hi16 * 8 + v;
            out[((size_t)b * SEQ + s) * DIM + h * HDIM + n] = o[f][v] * inv[v];
        }
    }
}

// ---------------------------------------------------------------------------
extern "C" void kernel_launch(void* const* d_in, const int* in_sizes, int n_in,
                              void* d_out, int out_size, void* d_ws, size_t ws_size,
                              hipStream_t stream) {
    (void)in_sizes; (void)n_in; (void)out_size; (void)ws_size;
    const float* x  = (const float*)d_in[0];
    const float* Wq = (const float*)d_in[1];
    const float* bq = (const float*)d_in[2];
    const float* Wk = (const float*)d_in[3];
    const float* bk = (const float*)d_in[4];
    const float* Wv = (const float*)d_in[5];
    const float* bv = (const float*)d_in[6];
    float* out = (float*)d_out;

    const size_t qkv_elems = (size_t)BATCH * HEADS * SEQ * HDIM; // 4M f16 = 8MB each
    _Float16* Qf  = (_Float16*)d_ws;
    _Float16* Kf  = Qf + qkv_elems;
    _Float16* Vtf = Kf + qkv_elems;   // stored transposed [B,H,dh,S]

    dim3 gProj(MTOT / 64, DIM / 128, 3);
    qkv_proj_kernel<<<gProj, 256, 0, stream>>>(x, Wq, bq, Wk, bk, Wv, bv, Qf, Kf, Vtf);

    dim3 gAttn(SEQ / 128, BATCH * HEADS);
    attn_kernel<<<gAttn, 256, 0, stream>>>(Qf, Kf, Vtf, out);
}
